// TemporalTransformer_18107582120301
// MI455X (gfx1250) — matrix operand order repaired and checked by reading, not compiled, not run on hardware
//
#include <hip/hip_runtime.h>
#include <hip/hip_bf16.h>
#include <math.h>

// ---- CDNA5 feature detection (device pass only; host pass falls back) ------
#if defined(__AMDGCN__) && __has_builtin(__builtin_amdgcn_global_load_async_to_lds_b128) && __has_builtin(__builtin_amdgcn_s_wait_asynccnt)
#define USE_ASYNC 1
#else
#define USE_ASYNC 0
#endif

#if defined(__AMDGCN__) && __has_builtin(__builtin_amdgcn_tensor_load_to_lds) && __has_builtin(__builtin_amdgcn_s_wait_tensorcnt)
#define USE_TDM 1
#else
#define USE_TDM 0
#endif

#if USE_ASYNC || USE_TDM
// async-to-LDS builtin takes pointers to 16-byte int vectors (per hipcc
// diagnostic: "__attribute__((__vector_size__(4*sizeof(int)))) int __device__ *")
typedef int v4i_b128 __attribute__((vector_size(16)));
#define AS1P(p) ((__attribute__((address_space(1))) v4i_b128*)(void*)(p))
#define AS3P(p) ((__attribute__((address_space(3))) v4i_b128*)(void*)(p))
#endif

namespace {

constexpr int LAY = 4;
constexpr int Dm  = 512;
constexpr int Hn  = 8;
constexpr int HDm = 64;
constexpr int FFm = 2048;
constexpr int Bb  = 4;
constexpr int Ss  = 2048;
constexpr int Mm  = Bb * Ss;   // 8192 rows

typedef __bf16 v16bf __attribute__((ext_vector_type(16)));
typedef float  v8f   __attribute__((ext_vector_type(8)));
#if USE_TDM
typedef unsigned int u32x4 __attribute__((ext_vector_type(4)));
typedef int          i32x4 __attribute__((ext_vector_type(4)));
typedef int          i32x8 __attribute__((ext_vector_type(8)));
#endif

union Frag16 { v16bf v; uint4 u[2]; };

__device__ __forceinline__ unsigned short f2bf(float f) {
  union { float f; unsigned u; } x; x.f = f;
  unsigned r = x.u + 0x7FFFu + ((x.u >> 16) & 1u);  // round-to-nearest-even
  return (unsigned short)(r >> 16);
}

__device__ __forceinline__ float geluf(float x) {
  return 0.5f * x * (1.0f + erff(x * 0.70710678118654752440f));
}

#if USE_TDM
__device__ __forceinline__ unsigned lds_byte_offset(const void* p) {
  return (unsigned)(unsigned long long)(__attribute__((address_space(3))) const char*)p;
}
#endif

// ---------------------------------------------------------------------------
// GEMM: C[M,N] = A_bf16[M,K] @ W_f32->bf16[K,N] + bias, optional GELU.
// OUT_MODE: 0 = bf16 row-major, 1 = bf16 scatter to [B,H,S,HD], 2 = f32 +=
// Block 256 thr (8 waves), tile 128x128x32; wave = 64x32 = 8 WMMA accums.
// A-tile staged by the Tensor Data Mover (one D# per k-step, wave 0 issues).
// ---------------------------------------------------------------------------
template<int OUT_MODE, int ACT>
__global__ __launch_bounds__(256) void
gemm_bf16_wmma(const unsigned short* __restrict__ A,
               const float* __restrict__ W,
               const float* __restrict__ bias,
               void* __restrict__ out,
               int N, int K)
{
  __shared__ alignas(16) unsigned short Alds[128][32];  // [m][k]
  __shared__ alignas(16) unsigned short Blds[128][32];  // [n][k] (transposed)

  const int tid  = threadIdx.x;
  const int lane = tid & 31;
  const int wid  = tid >> 5;
  const int l15  = lane & 15;
  const int g    = lane >> 4;
  const int wm   = wid & 1;     // 64-row slice
  const int wn   = wid >> 1;    // 32-col slice

  const int m0 = blockIdx.y * 128;
  const int n0 = blockIdx.x * 128;

  v8f acc[4][2];
  #pragma unroll
  for (int i = 0; i < 4; ++i)
    #pragma unroll
    for (int j = 0; j < 2; ++j)
      #pragma unroll
      for (int e = 0; e < 8; ++e) acc[i][j][e] = 0.0f;

  const int arow = tid >> 1, aseg = tid & 1;  // A staging fallback
  const int bk   = tid >> 3, bseg = tid & 7;  // B staging: 32 k x 8 col groups
  (void)arow; (void)aseg;

  for (int kt = 0; kt < K; kt += 32) {
#if USE_TDM
    if (wid == 0) {  // one TDM descriptor per block: 128x32 bf16 tile -> LDS
      const unsigned long long ga =
          (unsigned long long)(const void*)(A + (size_t)m0 * K + kt);
      u32x4 g0;
      g0[0] = 1u;                                        // count=1 (user D#)
      g0[1] = lds_byte_offset(&Alds[0][0]);              // lds_addr
      g0[2] = (unsigned)(ga & 0xFFFFFFFFu);              // global_addr[31:0]
      g0[3] = (unsigned)((ga >> 32) & 0x01FFFFFFu) | (2u << 30);  // addr hi | type=2
      i32x8 g1;
      g1[0] = 1 << 16;                                   // data_size=2B
      g1[1] = (K & 0xFFFF) << 16;                        // tensor_dim0 lo
      g1[2] = ((K >> 16) & 0xFFFF) | ((Mm & 0xFFFF) << 16);  // dim0 hi | dim1 lo
      g1[3] = ((Mm >> 16) & 0xFFFF) | (32 << 16);        // dim1 hi | tile_dim0=32
      g1[4] = 128;                                       // tile_dim1=128, tile_dim2=0
      g1[5] = K;                                         // tensor_dim0_stride lo
      g1[6] = 0;
      g1[7] = 0;
      i32x4 z4 = {0, 0, 0, 0};
#if __clang_major__ >= 23
      i32x8 z8 = {0, 0, 0, 0, 0, 0, 0, 0};
      __builtin_amdgcn_tensor_load_to_lds(g0, g1, z4, z4, z8, 0);
#else
      __builtin_amdgcn_tensor_load_to_lds(g0, g1, z4, z4, 0);
#endif
      __builtin_amdgcn_s_wait_tensorcnt(0);
    }
#elif USE_ASYNC
    { // per-lane async global->LDS (no VGPR round-trip)
      const unsigned short* gsrc = A + (size_t)(m0 + arow) * K + kt + aseg * 16;
      __builtin_amdgcn_global_load_async_to_lds_b128(
          AS1P(gsrc), AS3P(&Alds[arow][aseg * 16]), 0, 0);
      __builtin_amdgcn_global_load_async_to_lds_b128(
          AS1P(gsrc + 8), AS3P(&Alds[arow][aseg * 16 + 8]), 0, 0);
    }
#else
    { // stage A (already bf16): 16 elems = 2x b128 per thread
      const uint4* src = (const uint4*)(A + (size_t)(m0 + arow) * K + kt + aseg * 16);
      *(uint4*)&Alds[arow][aseg * 16]     = src[0];
      *(uint4*)&Alds[arow][aseg * 16 + 8] = src[1];
    }
#endif
    { // stage B: f32 -> bf16 convert-on-the-fly, transpose to [n][k]
      const float4* ws = (const float4*)(W + (size_t)(kt + bk) * N + n0 + bseg * 16);
      #pragma unroll
      for (int c = 0; c < 4; ++c) {
        float4 f = ws[c];
        Blds[bseg * 16 + c * 4 + 0][bk] = f2bf(f.x);
        Blds[bseg * 16 + c * 4 + 1][bk] = f2bf(f.y);
        Blds[bseg * 16 + c * 4 + 2][bk] = f2bf(f.z);
        Blds[bseg * 16 + c * 4 + 3][bk] = f2bf(f.w);
      }
    }
    if (kt + 32 < K)  // next-tile prefetch hint -> global_prefetch_b8
      __builtin_prefetch(W + (size_t)(kt + 32 + bk) * N + n0, 0, 0);
#if USE_ASYNC && !USE_TDM
    __builtin_amdgcn_s_wait_asynccnt(0);
#endif
    __syncthreads();

    // A-frag: lanes 0-15 carry K 0-7 & 16-23, lanes 16-31 shifted by 8 (ISA 7.12.2)
    Frag16 ag[4], bg[2];
    #pragma unroll
    for (int am = 0; am < 4; ++am) {
      const int r = wm * 64 + am * 16 + l15;
      ag[am].u[0] = *(const uint4*)&Alds[r][g * 8];
      ag[am].u[1] = *(const uint4*)&Alds[r][16 + g * 8];
    }
    // B-frag: [n][k] layout -> same K order as A (K 0-7 & 16-23, lanes16-31 shifted by 8)
    #pragma unroll
    for (int bn = 0; bn < 2; ++bn) {
      const int c = wn * 32 + bn * 16 + l15;
      bg[bn].u[0] = *(const uint4*)&Blds[c][g * 8];
      bg[bn].u[1] = *(const uint4*)&Blds[c][16 + g * 8];
    }
    #pragma unroll
    for (int am = 0; am < 4; ++am)
      #pragma unroll
      for (int bn = 0; bn < 2; ++bn)
        acc[am][bn] = __builtin_amdgcn_wmma_f32_16x16x32_bf16(
            false, ag[am].v, false, bg[bn].v, (short)0, acc[am][bn], false, false);

    __syncthreads();
  }

  // epilogue: C-frag lane mapping (lane%16 = N, vgpr + 8*(lane/16) = M)
  #pragma unroll
  for (int am = 0; am < 4; ++am) {
    #pragma unroll
    for (int bn = 0; bn < 2; ++bn) {
      const int col = n0 + wn * 32 + bn * 16 + l15;
      const float bv = bias[col];
      #pragma unroll
      for (int v = 0; v < 8; ++v) {
        const int row = m0 + wm * 64 + am * 16 + v + 8 * g;
        float val = acc[am][bn][v] + bv;
        if (ACT == 1) val = geluf(val);
        if (OUT_MODE == 0) {
          ((unsigned short*)out)[(size_t)row * N + col] = f2bf(val);
        } else if (OUT_MODE == 1) {
          const int b = row / Ss, s = row % Ss;
          const int h = col / HDm, hd = col % HDm;
          ((unsigned short*)out)[(((size_t)b * Hn + h) * Ss + s) * HDm + hd] = f2bf(val);
        } else {
          ((float*)out)[(size_t)row * N + col] += val;
        }
      }
    }
  }
}

// ---------------------------------------------------------------------------
// Flash attention (ALiBi + key-padding mask), 1 wave per 16 q rows.
// Block 128 thr = 4 waves = 64 q rows. K/V tiles 64x64 bf16 through LDS.
// K tile staged with GLOBAL_LOAD_ASYNC_TO_LDS (ASYNCcnt), V transposed on VALU.
// ---------------------------------------------------------------------------
__global__ __launch_bounds__(128) void
flash_attn_wmma(const unsigned short* __restrict__ Q,
                const unsigned short* __restrict__ Kb,
                const unsigned short* __restrict__ Vb,
                const unsigned char* __restrict__ mask,
                unsigned short* __restrict__ Ctx)
{
  __shared__ alignas(16) unsigned short Klds[64][64];      // [j][d] (B_lds[n][k] for QK^T)
  __shared__ alignas(16) unsigned short Vtld[64][64];      // [d][j] (B_lds[n][k] for P@V)
  __shared__ alignas(16) unsigned short Plds[4][16][64];   // per-wave P (C->A layout swap)

  const int tid  = threadIdx.x;
  const int lane = tid & 31, wid = tid >> 5;
  const int l15  = lane & 15, g = lane >> 4;
  const int bh = blockIdx.y;
  const int b = bh / Hn, h = bh % Hn;
  const int qb = blockIdx.x * 64 + wid * 16;

  const float slope   = exp2f(-(float)(h + 1));  // H=8 -> slopes 2^-1..2^-8
  const float inv_sm1 = 1.0f / (float)(Ss - 1);
  const float scale   = 0.125f;                  // HD^-0.5

  const unsigned short* qp = Q  + (size_t)bh * Ss * HDm;
  const unsigned short* kp = Kb + (size_t)bh * Ss * HDm;
  const unsigned short* vp = Vb + (size_t)bh * Ss * HDm;

  Frag16 qf[2];
  #pragma unroll
  for (int ks = 0; ks < 2; ++ks) {   // Q 16x64 = two A-frags, loaded from global
    const unsigned short* r = qp + (size_t)(qb + l15) * HDm + ks * 32;
    qf[ks].u[0] = *(const uint4*)(r + g * 8);
    qf[ks].u[1] = *(const uint4*)(r + 16 + g * 8);
  }

  float mrow[8], lrow[8];
  v8f ctx[4];
  #pragma unroll
  for (int v = 0; v < 8; ++v) { mrow[v] = -INFINITY; lrow[v] = 0.0f; }
  #pragma unroll
  for (int nb = 0; nb < 4; ++nb)
    #pragma unroll
    for (int e = 0; e < 8; ++e) ctx[nb][e] = 0.0f;

  const int sj = tid >> 1, shf = tid & 1;

  for (int kb0 = 0; kb0 < Ss; kb0 += 64) {
    { // stage K tile (async path: direct global->LDS, no VGPR data)
      const unsigned short* gk = kp + (size_t)(kb0 + sj) * HDm + shf * 32;
#if USE_ASYNC
      #pragma unroll
      for (int c = 0; c < 4; ++c)
        __builtin_amdgcn_global_load_async_to_lds_b128(
            AS1P(gk + c * 8), AS3P(&Klds[sj][shf * 32 + c * 8]), 0, 0);
#else
      const uint4* ks4 = (const uint4*)gk;
      uint4* kd = (uint4*)&Klds[sj][shf * 32];
      kd[0] = ks4[0]; kd[1] = ks4[1]; kd[2] = ks4[2]; kd[3] = ks4[3];
#endif
      // stage transposed V tile (transpose must go through VGPRs)
      uint4 vv[4];
      const uint4* vs4 = (const uint4*)(vp + (size_t)(kb0 + sj) * HDm + shf * 32);
      vv[0] = vs4[0]; vv[1] = vs4[1]; vv[2] = vs4[2]; vv[3] = vs4[3];
      const unsigned short* ve = (const unsigned short*)vv;
      #pragma unroll
      for (int dd = 0; dd < 32; ++dd) Vtld[shf * 32 + dd][sj] = ve[dd];
    }
#if USE_ASYNC
    __builtin_amdgcn_s_wait_asynccnt(0);
#endif
    __syncthreads();

    // scores S = Q K^T : 16x64 in 4 f32 C-frags
    v8f sc[4];
    #pragma unroll
    for (int jb = 0; jb < 4; ++jb) {
      #pragma unroll
      for (int e = 0; e < 8; ++e) sc[jb][e] = 0.0f;
      #pragma unroll
      for (int ks = 0; ks < 2; ++ks) {
        Frag16 kf;
        kf.u[0] = *(const uint4*)&Klds[jb * 16 + l15][ks * 32 + g * 8];
        kf.u[1] = *(const uint4*)&Klds[jb * 16 + l15][ks * 32 + 16 + g * 8];
        sc[jb] = __builtin_amdgcn_wmma_f32_16x16x32_bf16(
            false, qf[ks].v, false, kf.v, (short)0, sc[jb], false, false);
      }
    }

    bool mk[4];
    #pragma unroll
    for (int jb = 0; jb < 4; ++jb)
      mk[jb] = mask[(size_t)b * Ss + kb0 + jb * 16 + l15] != 0;

    // scale + alibi + mask, online softmax (row = vgpr + 8*(lane/16))
    float alpha[8];
    #pragma unroll
    for (int v = 0; v < 8; ++v) {
      const int qrow = qb + v + 8 * g;
      float mx = -INFINITY;
      #pragma unroll
      for (int jb = 0; jb < 4; ++jb) {
        const int j = kb0 + jb * 16 + l15;
        float s = sc[jb][v] * scale + slope * (float)(qrow - j) * inv_sm1;
        if (mk[jb]) s = -1e30f;
        sc[jb][v] = s;
        mx = fmaxf(mx, s);
      }
      #pragma unroll
      for (int off = 8; off; off >>= 1)
        mx = fmaxf(mx, __shfl_xor(mx, off, 16));   // row lives across 16 lanes
      const float mnew = fmaxf(mrow[v], mx);
      alpha[v] = __expf(mrow[v] - mnew);
      float rs = 0.0f;
      #pragma unroll
      for (int jb = 0; jb < 4; ++jb) {
        float p = mk[jb] ? 0.0f : __expf(sc[jb][v] - mnew);
        sc[jb][v] = p;
        rs += p;
      }
      #pragma unroll
      for (int off = 8; off; off >>= 1)
        rs += __shfl_xor(rs, off, 16);
      lrow[v] = lrow[v] * alpha[v] + rs;
      mrow[v] = mnew;
    }
    #pragma unroll
    for (int nb = 0; nb < 4; ++nb)
      #pragma unroll
      for (int v = 0; v < 8; ++v) ctx[nb][v] *= alpha[v];

    // P: C-layout -> LDS -> A-layout (bf16)
    #pragma unroll
    for (int v = 0; v < 8; ++v)
      #pragma unroll
      for (int jb = 0; jb < 4; ++jb)
        Plds[wid][v + 8 * g][jb * 16 + l15] = f2bf(sc[jb][v]);

    // ctx += P @ V
    #pragma unroll
    for (int ks = 0; ks < 2; ++ks) {
      Frag16 pf;
      const unsigned short* pr = &Plds[wid][l15][ks * 32];
      pf.u[0] = *(const uint4*)(pr + g * 8);
      pf.u[1] = *(const uint4*)(pr + 16 + g * 8);
      #pragma unroll
      for (int nb = 0; nb < 4; ++nb) {
        Frag16 vf;
        vf.u[0] = *(const uint4*)&Vtld[nb * 16 + l15][ks * 32 + g * 8];
        vf.u[1] = *(const uint4*)&Vtld[nb * 16 + l15][ks * 32 + 16 + g * 8];
        ctx[nb] = __builtin_amdgcn_wmma_f32_16x16x32_bf16(
            false, pf.v, false, vf.v, (short)0, ctx[nb], false, false);
      }
    }
    __syncthreads();
  }

  // normalize (l==0 => fully masked batch row => 0, matches reference)
  #pragma unroll
  for (int v = 0; v < 8; ++v) {
    const float inv  = lrow[v] > 0.0f ? 1.0f / lrow[v] : 0.0f;
    const int   qrow = qb + v + 8 * g;
    #pragma unroll
    for (int nb = 0; nb < 4; ++nb) {
      const int d = nb * 16 + l15;
      Ctx[((size_t)b * Ss + qrow) * Dm + h * HDm + d] = f2bf(ctx[nb][v] * inv);
    }
  }
}

// ---------------------------------------------------------------------------
// LayerNorm: one wave per row of D=512 (16 f32/lane), shfl_xor reductions.
// ---------------------------------------------------------------------------
template<int BF16OUT>
__global__ __launch_bounds__(256) void
layernorm_k(const float* __restrict__ x, const float* __restrict__ w,
            const float* __restrict__ bsh, void* __restrict__ out)
{
  const int tid = threadIdx.x, lane = tid & 31, wid = tid >> 5;
  const int row = blockIdx.x * 8 + wid;
  const float* xr = x + (size_t)row * Dm;
  float v[16];
  float s = 0.f;
  #pragma unroll
  for (int i = 0; i < 16; ++i) { v[i] = xr[lane + i * 32]; s += v[i]; }
  #pragma unroll
  for (int off = 16; off; off >>= 1) s += __shfl_xor(s, off);
  const float mean = s * (1.0f / Dm);
  float q = 0.f;
  #pragma unroll
  for (int i = 0; i < 16; ++i) { float d = v[i] - mean; q += d * d; }
  #pragma unroll
  for (int off = 16; off; off >>= 1) q += __shfl_xor(q, off);
  const float rs = rsqrtf(q * (1.0f / Dm) + 1e-5f);
  #pragma unroll
  for (int i = 0; i < 16; ++i) {
    const int c = lane + i * 32;
    const float y = (v[i] - mean) * rs * w[c] + bsh[c];
    if (BF16OUT) ((unsigned short*)out)[(size_t)row * Dm + c] = f2bf(y);
    else         ((float*)out)[(size_t)row * Dm + c] = y;
  }
}

__global__ void copy_f32(const float* __restrict__ src, float* __restrict__ dst, int n) {
  const int i = blockIdx.x * blockDim.x + threadIdx.x;
  if (i < n) dst[i] = src[i];
}

} // namespace

extern "C" void kernel_launch(void* const* d_in, const int* in_sizes, int n_in,
                              void* d_out, int out_size, void* d_ws, size_t ws_size,
                              hipStream_t stream) {
  (void)in_sizes; (void)n_in; (void)out_size; (void)ws_size;
  const float* x            = (const float*)d_in[0];
  const unsigned char* mask = (const unsigned char*)d_in[1];
  const float* ln1_w = (const float*)d_in[2];
  const float* ln1_b = (const float*)d_in[3];
  const float* Wq = (const float*)d_in[4];
  const float* bq = (const float*)d_in[5];
  const float* Wk = (const float*)d_in[6];
  const float* bk = (const float*)d_in[7];
  const float* Wv = (const float*)d_in[8];
  const float* bv = (const float*)d_in[9];
  const float* Wo = (const float*)d_in[10];
  const float* bo = (const float*)d_in[11];
  const float* ln2_w = (const float*)d_in[12];
  const float* ln2_b = (const float*)d_in[13];
  const float* W1 = (const float*)d_in[14];
  const float* b1 = (const float*)d_in[15];
  const float* W2 = (const float*)d_in[16];
  const float* b2 = (const float*)d_in[17];
  const float* fn_w = (const float*)d_in[18];
  const float* fn_b = (const float*)d_in[19];

  // workspace layout (bytes)
  char* ws = (char*)d_ws;
  float*          resid = (float*)ws;                          // M*D f32   (16.78 MB)
  unsigned short* hbuf  = (unsigned short*)(ws + 16777216);    // M*D bf16
  unsigned short* qbuf  = (unsigned short*)(ws + 25165824);    // [B,H,S,HD] bf16
  unsigned short* kbuf  = (unsigned short*)(ws + 33554432);
  unsigned short* vbuf  = (unsigned short*)(ws + 41943040);
  unsigned short* cbuf  = (unsigned short*)(ws + 50331648);    // ctx [M,D] bf16
  unsigned short* fbuf  = (unsigned short*)(ws + 58720256);    // [M,FF] bf16 (33.55 MB)

  copy_f32<<<(Mm * Dm) / 256, 256, 0, stream>>>(x, resid, Mm * Dm);

  const dim3 g512(Dm / 128, Mm / 128);   // (4, 64)
  const dim3 gff (FFm / 128, Mm / 128);  // (16, 64)
  const dim3 gfa (Ss / 64, Bb * Hn);     // (32, 32)

  for (int i = 0; i < LAY; ++i) {
    layernorm_k<1><<<Mm / 8, 256, 0, stream>>>(resid, ln1_w + i * Dm, ln1_b + i * Dm, hbuf);
    gemm_bf16_wmma<1, 0><<<g512, 256, 0, stream>>>(hbuf, Wq + (size_t)i * Dm * Dm, bq + i * Dm, qbuf, Dm, Dm);
    gemm_bf16_wmma<1, 0><<<g512, 256, 0, stream>>>(hbuf, Wk + (size_t)i * Dm * Dm, bk + i * Dm, kbuf, Dm, Dm);
    gemm_bf16_wmma<1, 0><<<g512, 256, 0, stream>>>(hbuf, Wv + (size_t)i * Dm * Dm, bv + i * Dm, vbuf, Dm, Dm);
    flash_attn_wmma<<<gfa, 128, 0, stream>>>(qbuf, kbuf, vbuf, mask, cbuf);
    gemm_bf16_wmma<2, 0><<<g512, 256, 0, stream>>>(cbuf, Wo + (size_t)i * Dm * Dm, bo + i * Dm, resid, Dm, Dm);
    layernorm_k<1><<<Mm / 8, 256, 0, stream>>>(resid, ln2_w + i * Dm, ln2_b + i * Dm, hbuf);
    gemm_bf16_wmma<0, 1><<<gff, 256, 0, stream>>>(hbuf, W1 + (size_t)i * Dm * FFm, b1 + i * FFm, fbuf, FFm, Dm);
    gemm_bf16_wmma<2, 0><<<g512, 256, 0, stream>>>(fbuf, W2 + (size_t)i * FFm * Dm, b2 + i * Dm, resid, Dm, FFm);
  }
  layernorm_k<0><<<Mm / 8, 256, 0, stream>>>(resid, fn_w, fn_b, d_out);
}
